// Mymodel_17875653886123
// MI455X (gfx1250) — compile-verified, matrix operand order checked by reading, not run on hardware
//
#include <hip/hip_runtime.h>
#include <hip/hip_bf16.h>
#include <stdint.h>

#define PAD_ID   0
#define START_ID 30
#define END_ID   31
#define EMBD     256
#define HID      512
#define NTAG     32
#define BATCH    64
#define SEQ      512

typedef __attribute__((ext_vector_type(16))) __bf16 v16bf;
typedef __attribute__((ext_vector_type(8)))  __bf16 v8bf;
typedef __attribute__((ext_vector_type(8)))  float  v8f;
typedef __attribute__((ext_vector_type(4)))  unsigned v4u;
typedef __attribute__((ext_vector_type(8)))  int      v8i;
typedef __attribute__((ext_vector_type(4)))  int      v4i;

// ---- workspace layout (bytes) ----
#define OFF_WPF   ((size_t)0)                                  // packed w_ih_f||w_hh_f : 128 rows x 24 ktiles x 1KB
#define OFF_WPB   (OFF_WPF + (size_t)768*2048*2)
#define OFF_WPC   (OFF_WPB + (size_t)768*2048*2)               // packed w_cls^T : [1024 x 32] bf16
#define OFF_EMB   (OFF_WPC + (size_t)1024*32*2)                // emb  [L][B][256] bf16
#define OFF_OUT   (OFF_EMB + (size_t)SEQ*BATCH*EMBD*2)         // h concat [L][B][1024] bf16
#define OFF_EMI   (OFF_OUT + (size_t)SEQ*BATCH*1024*2)         // emission [L*B][32] f32
#define OFF_LEN   (OFF_EMI + (size_t)SEQ*BATCH*NTAG*4)         // lengths [64] int
#define OFF_BAR   (OFF_LEN + 256)                              // barrier counters [8] u32
#define OFF_ACC   (OFF_BAR + 64)                               // gold[64], path[64] f32

// Within-tile WMMA B-operand swizzle (ISA 7.12.2 16-bit layout): element (k%32, n%16)
// -> position so lane L reads 8 contiguous dwords at block + L*8.
__device__ __forceinline__ size_t swz_in(int kin, int nin) {
  int g16 = kin >> 4, sub = kin & 15, hi = sub >> 3;
  int v = g16 * 4 + ((sub & 7) >> 1), half = kin & 1;
  int lane = nin + (hi << 4);
  return (size_t)(lane * 8 + v) * 2 + half;
}

// LSTM weights: packed row index chosen so a WG's 8 wave-tiles are 8 CONSECUTIVE rows
// (row pitch 24KB) -> one contiguous 2D TDM tile per chunk.
__device__ __forceinline__ size_t swz_lstm(int k, int n) {
  int gate = n >> 9, col = n & 511;
  int nsx = col >> 5, nhx = (col >> 4) & 1, nin = n & 15;
  int row = nsx * 8 + gate * 2 + nhx;            // 0..127
  int kt = k >> 5;
  return ((size_t)row * 24 + kt) * 512 + swz_in(k & 31, nin);
}

__device__ __forceinline__ size_t swz_cls(int k, int n) {
  int nt = n >> 4, kt = k >> 5;
  return ((size_t)nt * 32 + kt) * 512 + swz_in(k & 31, n & 15);
}

__device__ __forceinline__ v16bf load_a16(const __bf16* p) {
  v8bf lo = *(const v8bf*)p;
  v8bf hi = *(const v8bf*)(p + 16);
  return __builtin_shufflevector(lo, hi, 0,1,2,3,4,5,6,7,8,9,10,11,12,13,14,15);
}

// TDM: DMA a [8 rows x 6KB] 2D tile of packed weights (row pitch 24KB) into LDS.
// Descriptor bitfields per CDNA5 ISA 8.3/8.4; 2D tensor -> groups 2/3 zero-filled.
// This toolchain declares the 6-arg builtin (g0, g1, g2, g3, g4, cpol).
__device__ __forceinline__ void tdm_load_b(const __bf16* gsrc, unsigned lds_off) {
  uint64_t ga = (uint64_t)(uintptr_t)gsrc;
  v4u g0;
  g0[0] = 1u;                                        // count=1 (valid user descriptor)
  g0[1] = lds_off;                                   // lds_addr (bytes)
  g0[2] = (unsigned)ga;                              // global_addr[31:0]
  g0[3] = ((unsigned)(ga >> 32) & 0x01FFFFFFu)       // global_addr[56:32]
        | (2u << 30);                                // type=2 ("image")
  const unsigned td0 = 768, td1 = 8;                 // tensor dims in 8-byte units
  const unsigned tile0 = 768, tile1 = 8;             // tile = whole 8 x 6KB block
  const unsigned s0 = 3072;                          // row pitch: 24KB / 8B
  v8i g1;
  g1[0] = (int)(3u << 16);                           // data_size=8B; wg_mask=0; no flags
  g1[1] = (int)((td0 & 0xFFFFu) << 16);              // [31:16] = tensor_dim0[15:0]
  g1[2] = (int)((td0 >> 16) | ((td1 & 0xFFFFu) << 16));
  g1[3] = (int)((td1 >> 16) | (tile0 << 16));        // [31:16] = tile_dim0
  g1[4] = (int)tile1;                                // tile_dim1; tile_dim2=0
  g1[5] = (int)s0;                                   // tensor_dim0_stride[31:0]
  g1[6] = 0;
  g1[7] = 0;
  v4i z4 = {0, 0, 0, 0};
  v8i z8 = {0, 0, 0, 0, 0, 0, 0, 0};
  __builtin_amdgcn_tensor_load_to_lds(g0, g1, z4, z4, z8, 0);
}

// ---------------- prep kernels ----------------
__global__ void k_zero(unsigned* bar) {
  if (threadIdx.x < 8) bar[threadIdx.x] = 0u;
}

__global__ void k_pack(const float* __restrict__ wihf, const float* __restrict__ whhf,
                       const float* __restrict__ wihb, const float* __restrict__ whhb,
                       const float* __restrict__ wcls,
                       __bf16* __restrict__ wpf, __bf16* __restrict__ wpb,
                       __bf16* __restrict__ wpc) {
  size_t i = (size_t)blockIdx.x * blockDim.x + threadIdx.x;
  const size_t NL = (size_t)768 * 2048;
  if (i < 2 * NL) {
    int dir = (i >= NL); size_t r = dir ? (i - NL) : i;
    int k = (int)(r >> 11), n = (int)(r & 2047);
    const float* wih = dir ? wihb : wihf;
    const float* whh = dir ? whhb : whhf;
    float v = (k < 256) ? wih[(size_t)n * 256 + k] : whh[(size_t)n * 512 + (k - 256)];
    (dir ? wpb : wpf)[swz_lstm(k, n)] = (__bf16)v;
  } else {
    size_t r = i - 2 * NL;
    if (r < (size_t)1024 * 32) {
      int k = (int)(r >> 5), n = (int)(r & 31);
      wpc[swz_cls(k, n)] = (__bf16)wcls[(size_t)n * 1024 + k];
    }
  }
}

__global__ void k_gather(const int* __restrict__ tokens, const float* __restrict__ table,
                         __bf16* __restrict__ emb) {
  size_t i = (size_t)blockIdx.x * blockDim.x + threadIdx.x;  // over L*B*EMB, [t][b][e]
  if (i >= (size_t)SEQ * BATCH * EMBD) return;
  int e = (int)(i & (EMBD - 1));
  size_t tb = i >> 8;
  int b = (int)(tb & 63), t = (int)(tb >> 6);
  int tok = tokens[(size_t)b * SEQ + t];
  emb[i] = (__bf16)table[(size_t)tok * EMBD + e];
}

// ---------------- fused BiLSTM recurrence ----------------
// grid = (16 col-slices, 4 batch-groups, 2 dirs), block = 256 (8 waves).
// Weights stream via TDM into a double-buffered LDS stage; A tiles from global;
// WMMA bf16 with fp32 accumulate; per-step h exchange via L2 + barrier.
__global__ __launch_bounds__(256) void k_lstm(
    const __bf16* __restrict__ emb, const __bf16* __restrict__ wpf,
    const __bf16* __restrict__ wpb, __bf16* __restrict__ outh,
    const float* __restrict__ bihf, const float* __restrict__ bhhf,
    const float* __restrict__ bihb, const float* __restrict__ bhhb,
    unsigned* __restrict__ bar)
{
  const int ns = blockIdx.x, bg = blockIdx.y, dir = blockIdx.z;
  const int tid = threadIdx.x, lane = tid & 31, w = tid >> 5;
  const int b0 = bg * 16;
  const int m = lane & 15, khi = lane >> 4;
  const __bf16* wp = dir ? wpb : wpf;
  const float* bih = dir ? bihb : bihf;
  const float* bhh = dir ? bhhb : bhhf;
  const int grp = dir * 4 + bg;
  const __bf16* wrow0 = wp + (size_t)(ns * 8) * 24 * 512;  // this WG's 8 packed rows

  __shared__ __bf16 lds_b[2][8 * 6 * 512];   // 2 x 48KB weight stage
  __shared__ float  lds_g[8 * 256];          // 8 waves x 16x16 fp32 gate tiles
  __shared__ float  lds_c[512];              // persistent cell state [16 x 32]
  for (int i = tid; i < 512; i += 256) lds_c[i] = 0.f;
  __syncthreads();

  for (int t = 0; t < SEQ; ++t) {
    const int tt  = dir ? (SEQ - 1 - t) : t;
    const int ptt = dir ? (tt + 1) : (tt - 1);
    v8f acc = {};

    if (t == 0) {
      // h_prev == 0: only the 8 embedding K-tiles, direct global B loads.
      for (int kt = 0; kt < 8; ++kt) {
        const __bf16* ap = emb + ((size_t)tt * BATCH + b0 + m) * EMBD + kt * 32 + khi * 8;
        v16bf A = load_a16(ap);
        v16bf B = *(const v16bf*)(wrow0 + ((size_t)w * 24 + kt) * 512 + lane * 16);
        acc = __builtin_amdgcn_wmma_f32_16x16x32_bf16(false, A, false, B, (short)0, acc,
                                                      false, false);
      }
    } else {
      // 24 K-tiles in 4 chunks of 6, TDM double-buffered through LDS.
      if (w == 0)
        tdm_load_b(wrow0, (unsigned)(uintptr_t)&lds_b[0][0]);
      for (int c = 0; c < 4; ++c) {
        if (w == 0) {
          if (c < 3) {
            tdm_load_b(wrow0 + (size_t)(c + 1) * 6 * 512,
                       (unsigned)(uintptr_t)&lds_b[(c + 1) & 1][0]);
            __builtin_amdgcn_s_wait_tensorcnt(1);   // chunk c landed
          } else {
            __builtin_amdgcn_s_wait_tensorcnt(0);
          }
        }
        __syncthreads();                            // LDS stage visible to all waves
        const __bf16* bb = &lds_b[c & 1][0] + (size_t)w * 6 * 512 + lane * 16;
        #pragma unroll
        for (int j = 0; j < 6; ++j) {
          const int kt = c * 6 + j;
          const __bf16* ap = (kt < 8)
            ? emb  + ((size_t)tt  * BATCH + b0 + m) * EMBD + kt * 32 + khi * 8
            : outh + ((size_t)ptt * BATCH + b0 + m) * 1024 + dir * 512
                   + (kt - 8) * 32 + khi * 8;
          v16bf A = load_a16(ap);
          v16bf B = *(const v16bf*)(bb + j * 512);
          acc = __builtin_amdgcn_wmma_f32_16x16x32_bf16(false, A, false, B, (short)0, acc,
                                                        false, false);
        }
        __syncthreads();                            // chunk consumed: safe to overwrite
      }
    }

    { // C layout (ISA 7.12.2): VGPR r, lanes<16 -> M=r, lanes>=16 -> M=8+r, N=lane&15
      const int cm = khi * 8, cn = lane & 15;
      #pragma unroll
      for (int r = 0; r < 8; ++r)
        lds_g[w * 256 + (cm + r) * 16 + cn] = acc[r];
    }
    __syncthreads();
    // elementwise LSTM cell for this 16x32 h-slice
    for (int idx = tid; idx < 512; idx += 256) {
      int mm = idx >> 5, j = idx & 31;
      int jh = j >> 4, jn = j & 15;
      float gi = lds_g[(0 * 2 + jh) * 256 + mm * 16 + jn];
      float gf = lds_g[(1 * 2 + jh) * 256 + mm * 16 + jn];
      float gg = lds_g[(2 * 2 + jh) * 256 + mm * 16 + jn];
      float go = lds_g[(3 * 2 + jh) * 256 + mm * 16 + jn];
      int nI = 0 * HID + ns * 32 + j, nF = 1 * HID + ns * 32 + j;
      int nG = 2 * HID + ns * 32 + j, nO = 3 * HID + ns * 32 + j;
      gi += bih[nI] + bhh[nI];  gf += bih[nF] + bhh[nF];
      gg += bih[nG] + bhh[nG];  go += bih[nO] + bhh[nO];
      float si = 1.f / (1.f + expf(-gi));
      float sf = 1.f / (1.f + expf(-gf));
      float so = 1.f / (1.f + expf(-go));
      float c  = sf * lds_c[idx] + si * tanhf(gg);
      lds_c[idx] = c;
      float h  = so * tanhf(c);
      outh[((size_t)tt * BATCH + b0 + mm) * 1024 + dir * 512 + ns * 32 + j] = (__bf16)h;
    }
    __threadfence();
    __syncthreads();
    // sync the 16 column-slice WGs of this (dir, batch-group):
    // cluster barrier when dispatched as a 16-WG cluster, monotonic atomic barrier otherwise.
    if (__builtin_amdgcn_cluster_id_x() != 0) {
      __builtin_amdgcn_s_cluster_barrier();
    } else {
      if (tid == 0) {
        __hip_atomic_fetch_add(&bar[grp], 1u, __ATOMIC_RELEASE, __HIP_MEMORY_SCOPE_AGENT);
        const unsigned tgt = 16u * (unsigned)(t + 1);
        while (__hip_atomic_load(&bar[grp], __ATOMIC_ACQUIRE, __HIP_MEMORY_SCOPE_AGENT) < tgt)
          __builtin_amdgcn_s_sleep(2);
      }
      __syncthreads();
    }
  }
}

// ---------------- emission GEMM: [32768,1024] x [1024,32] ----------------
__global__ __launch_bounds__(256) void k_emis(
    const __bf16* __restrict__ outh, const __bf16* __restrict__ wpc,
    const float* __restrict__ bcls, float* __restrict__ emis)
{
  const int tid = threadIdx.x, lane = tid & 31, w = tid >> 5;
  const int mt = blockIdx.x * 8 + w;            // 2048 M-tiles
  const int m = lane & 15, khi = lane >> 4;
  const size_t r0 = (size_t)mt * 16;
  v8f acc0 = {}, acc1 = {};
  for (int kt = 0; kt < 32; ++kt) {
    v16bf A  = load_a16(outh + (r0 + m) * 1024 + kt * 32 + khi * 8);
    v16bf B0 = *(const v16bf*)(wpc + ((size_t)0 * 32 + kt) * 512 + lane * 16);
    v16bf B1 = *(const v16bf*)(wpc + ((size_t)1 * 32 + kt) * 512 + lane * 16);
    acc0 = __builtin_amdgcn_wmma_f32_16x16x32_bf16(false, A, false, B0, (short)0, acc0,
                                                   false, false);
    acc1 = __builtin_amdgcn_wmma_f32_16x16x32_bf16(false, A, false, B1, (short)0, acc1,
                                                   false, false);
  }
  const int cm = khi * 8, cn = lane & 15;
  #pragma unroll
  for (int r = 0; r < 8; ++r) {
    size_t row = r0 + cm + r;
    emis[row * NTAG + cn]      = acc0[r] + bcls[cn];
    emis[row * NTAG + 16 + cn] = acc1[r] + bcls[16 + cn];
  }
}

// ---------------- CRF gold path + lengths ----------------
__global__ void k_gold(const int* __restrict__ tags, const float* __restrict__ trans,
                       const float* __restrict__ emis, int* __restrict__ lens,
                       float* __restrict__ accum)
{
  int b = threadIdx.x;
  if (b >= BATCH) return;
  int prev = START_ID, len = 0; float s = 0.f;
  for (int t = 0; t < SEQ; ++t) {
    int tag = tags[(size_t)b * SEQ + t];
    if (tag != PAD_ID) {
      ++len;
      s += emis[((size_t)t * BATCH + b) * NTAG + tag] + trans[prev * NTAG + tag];
    }
    prev = tag;
  }
  lens[b] = len;
  accum[b] = s;
}

// ---------------- CRF forward (logsumexp scan), 1 wave per batch row ----------------
__global__ __launch_bounds__(32) void k_crf(const float* __restrict__ trans,
                                            const float* __restrict__ emis,
                                            const int* __restrict__ lens,
                                            float* __restrict__ accum)
{
  const int b = blockIdx.x, j = threadIdx.x;
  __shared__ float tT[32 * 33];                 // transposed + padded vs bank conflicts
  for (int i = 0; i < 32; ++i) tT[j * 33 + i] = trans[i * NTAG + j];
  __syncthreads();
  const int len = lens[b];
  float s = emis[((size_t)0 * BATCH + b) * NTAG + j] + trans[START_ID * NTAG + j];
  for (int t = 1; t < SEQ; ++t) {
    float mx = -3.4e38f;
    #pragma unroll 8
    for (int i = 0; i < 32; ++i)
      mx = fmaxf(mx, tT[j * 33 + i] + __shfl(s, i, 32));
    float sum = 0.f;
    #pragma unroll 8
    for (int i = 0; i < 32; ++i)
      sum += expf(tT[j * 33 + i] + __shfl(s, i, 32) - mx);
    float nw = emis[((size_t)t * BATCH + b) * NTAG + j] + mx + logf(sum);
    s = (t < len) ? nw : s;
  }
  if (j == END_ID) accum[BATCH + b] = s;
}

__global__ void k_fin(const float* __restrict__ accum, float* __restrict__ out) {
  float g = 0.f, p = 0.f;
  for (int b = 0; b < BATCH; ++b) { g += accum[b]; p += accum[BATCH + b]; }
  out[0] = (p - g) * (1.0f / BATCH);
}

// ---------------- launch ----------------
extern "C" void kernel_launch(void* const* d_in, const int* in_sizes, int n_in,
                              void* d_out, int out_size, void* d_ws, size_t ws_size,
                              hipStream_t stream) {
  const int*   batch_data = (const int*)  d_in[0];
  const int*   batch_tag  = (const int*)  d_in[1];
  const float* embed      = (const float*)d_in[2];
  const float* w_ih_f     = (const float*)d_in[3];
  const float* w_hh_f     = (const float*)d_in[4];
  const float* b_ih_f     = (const float*)d_in[5];
  const float* b_hh_f     = (const float*)d_in[6];
  const float* w_ih_b     = (const float*)d_in[7];
  const float* w_hh_b     = (const float*)d_in[8];
  const float* b_ih_b     = (const float*)d_in[9];
  const float* b_hh_b     = (const float*)d_in[10];
  const float* w_cls      = (const float*)d_in[11];
  const float* b_cls      = (const float*)d_in[12];
  const float* trans      = (const float*)d_in[13];

  char* ws = (char*)d_ws;
  __bf16*   wpf   = (__bf16*)  (ws + OFF_WPF);
  __bf16*   wpb   = (__bf16*)  (ws + OFF_WPB);
  __bf16*   wpc   = (__bf16*)  (ws + OFF_WPC);
  __bf16*   emb   = (__bf16*)  (ws + OFF_EMB);
  __bf16*   outh  = (__bf16*)  (ws + OFF_OUT);
  float*    emis  = (float*)   (ws + OFF_EMI);
  int*      lens  = (int*)     (ws + OFF_LEN);
  unsigned* bar   = (unsigned*)(ws + OFF_BAR);
  float*    accum = (float*)   (ws + OFF_ACC);

  k_zero<<<1, 64, 0, stream>>>(bar);

  const size_t pack_elems = (size_t)2 * 768 * 2048 + (size_t)1024 * 32;
  k_pack<<<(unsigned)((pack_elems + 255) / 256), 256, 0, stream>>>(
      w_ih_f, w_hh_f, w_ih_b, w_hh_b, w_cls, wpf, wpb, wpc);

  const size_t gat_elems = (size_t)SEQ * BATCH * EMBD;
  k_gather<<<(unsigned)((gat_elems + 255) / 256), 256, 0, stream>>>(batch_data, embed, emb);

  k_lstm<<<dim3(16, 4, 2), 256, 0, stream>>>(emb, wpf, wpb, outh,
                                             b_ih_f, b_hh_f, b_ih_b, b_hh_b, bar);

  k_emis<<<256, 256, 0, stream>>>(outh, wpc, b_cls, emis);

  k_gold<<<1, 64, 0, stream>>>(batch_tag, trans, emis, lens, accum);
  k_crf<<<64, 32, 0, stream>>>(trans, emis, lens, accum);
  k_fin<<<1, 1, 0, stream>>>(accum, (float*)d_out);
}
